// re2graph_10754598109602
// MI455X (gfx1250) — compile-verified, matrix operand order checked by reading, not run on hardware
//
#include <hip/hip_runtime.h>

typedef __attribute__((ext_vector_type(16))) _Float16 v16h;
typedef __attribute__((ext_vector_type(8)))  float    v8f;

#define Bn   8
#define Sn   2048
#define Dn   256
#define BSn  (Bn * Sn)                       // 16384
#define BSDn ((long long)BSn * Dn)           // 4,194,304
#define BSSn ((long long)Bn * Sn * Sn)       // 33,554,432

__device__ __forceinline__ float lrelu(float x, float a) { return x > 0.f ? x : a * x; }

// ---------------------------------------------------------------- f32 -> f16
__global__ void re2g_cvt_f16(const float* __restrict__ x, _Float16* __restrict__ y, long long n) {
  long long stride = (long long)gridDim.x * blockDim.x;
  for (long long i = (long long)blockIdx.x * blockDim.x + threadIdx.x; i < n; i += stride)
    y[i] = (_Float16)x[i];
}

// ------------------------------------ f32 [K x N] -> f16 transposed [N x K], 256x256
__global__ void re2g_cvt_tr(const float* __restrict__ x, _Float16* __restrict__ y) {
  int idx = blockIdx.x * blockDim.x + threadIdx.x;   // 65536 threads
  int n = idx >> 8, k = idx & 255;
  y[n * 256 + k] = (_Float16)x[k * 256 + n];
}

// ---- async stage one 64x32-half tile (row-major src, row stride K halves) into LDS
__device__ __forceinline__ void stage_async(const _Float16* __restrict__ src, int K,
                                            int rowbase, int kk, _Float16* lds, int t) {
#pragma unroll
  for (int i = t; i < 256; i += 128) {
    int r = i >> 2, s = i & 3;
    unsigned la = (unsigned)(size_t)(lds + i * 8);
    int go = ((rowbase + r) * K + kk) * 2 + s * 16;   // byte offset
    asm volatile("global_load_async_to_lds_b128 %0, %1, %2"
                 :: "v"(la), "v"(go), "s"(src) : "memory");
  }
}

__device__ __forceinline__ v16h load_frag(const _Float16* base) {
  union { v16h v; uint4 q[2]; } u;
  u.q[0] = *(const uint4*)(base);        // K = kb + 0..7
  u.q[1] = *(const uint4*)(base + 16);   // K = kb + 16..23
  return u.v;
}

// ------------------------------------------------- generic batched WMMA GEMM
// C[b](MxN,f32-acc) = A[b](MxK f16, row-major) * BT[b](NxK f16, row-major)^T (+resid)
// K mult of 32. grid=(N/64, M/64, batch), block=128 (4 waves), 64x64 tile,
// double-buffered LDS staged via global_load_async_to_lds_b128.
// Optional outputs: Cf (f32), Ch (f16 row-major), Cht (f16 transposed [b][N][Sn],
// valid only for non-batched producer GEMMs whose rows are b*Sn+i).
__global__ void re2g_gemm_wmma(const _Float16* __restrict__ A, const _Float16* __restrict__ BT,
                               const float* __restrict__ resid, float* __restrict__ Cf,
                               _Float16* __restrict__ Ch, _Float16* __restrict__ Cht,
                               int K, int N, long long sA, long long sB, long long sC) {
  __shared__ __align__(16) _Float16 As[2][64 * 32];
  __shared__ __align__(16) _Float16 Bs[2][64 * 32];
  const int t = threadIdx.x;
  const int w = t >> 5, lane = t & 31;
  const int bz = blockIdx.z;
  const _Float16* Ab = A + (long long)bz * sA;
  const _Float16* Bb = BT + (long long)bz * sB;
  const long long cbase = (long long)bz * sC;
  const int row0 = blockIdx.y * 64;
  const int col0 = blockIdx.x * 64;
  const int m16 = lane & 15;
  const int kb = (lane < 16) ? 0 : 8;
  const int niter = K >> 5;

  v8f acc[4] = {};
  stage_async(Ab, K, row0, 0, As[0], t);
  stage_async(Bb, K, col0, 0, Bs[0], t);
  for (int it = 0; it < niter; ++it) {
    const int cur = it & 1, nxt = cur ^ 1;
    asm volatile("s_wait_asynccnt 0x0" ::: "memory");  // my async copies landed
    __syncthreads();                                   // everyone's landed / done reading nxt
    if (it + 1 < niter) {
      stage_async(Ab, K, row0, (it + 1) << 5, As[nxt], t);
      stage_async(Bb, K, col0, (it + 1) << 5, Bs[nxt], t);
    }
    const _Float16* arow = &As[cur][(w * 16 + m16) * 32 + kb];
    v16h af = load_frag(arow);
#pragma unroll
    for (int c = 0; c < 4; ++c) {
      v16h bf = load_frag(&Bs[cur][(c * 16 + m16) * 32 + kb]);
      acc[c] = __builtin_amdgcn_wmma_f32_16x16x32_f16(false, af, false, bf,
                                                      (short)0, acc[c], false, false);
    }
  }
  // C/D layout: vgpr v holds (M = v + (lane>=16 ? 8:0), N = lane%16)
  const int rowbase = row0 + w * 16 + ((lane < 16) ? 0 : 8);
#pragma unroll
  for (int c = 0; c < 4; ++c) {
    int col = col0 + c * 16 + m16;
#pragma unroll
    for (int v = 0; v < 8; ++v) {
      int rowg = rowbase + v;
      long long idx = cbase + (long long)rowg * N + col;
      float val = acc[c][v];
      if (resid) val += resid[idx];
      if (Cf) Cf[idx] = val;
      if (Ch) Ch[idx] = (_Float16)val;
      if (Cht) {  // transposed [b][N][Sn] with rows flattened as b*Sn+i
        long long bb = rowg >> 11;                 // / Sn
        long long ii = rowg & (Sn - 1);
        Cht[bb * ((long long)N * Sn) + (long long)col * Sn + ii] = (_Float16)val;
      }
    }
  }
}

// ------ per-row dot on transposed Wh: out[b*Sn+i] = sum_k WhT[b][k][i] * a[k]
__global__ void re2g_rowdot_t(const _Float16* __restrict__ WhT, const float* __restrict__ a,
                              float* __restrict__ out) {
  int i0 = blockIdx.x * 256;            // grid = BSn/256
  int b = i0 >> 11;                     // / Sn
  int i = (i0 & (Sn - 1)) + threadIdx.x;
  const _Float16* base = WhT + (long long)b * Dn * Sn + i;
  float s = 0.f;
#pragma unroll 8
  for (int k = 0; k < Dn; ++k) s += (float)base[(long long)k * Sn] * a[k];
  out[b * Sn + i] = s;
}

// -------------------- per-batch max of f2 over unmasked / masked positions
__global__ void re2g_batchmax(const float* __restrict__ f2, const float* __restrict__ mask,
                              float* __restrict__ mxU, float* __restrict__ mxM) {
  __shared__ float su[256], sm[256];
  int b = blockIdx.x, t = threadIdx.x;
  float u = -3.4e38f, m = -3.4e38f;
  for (int j = t; j < Sn; j += 256) {
    float v = f2[b * Sn + j];
    if (mask[b * Sn + j] > 0.f) u = fmaxf(u, v); else m = fmaxf(m, v);
  }
  su[t] = u; sm[t] = m; __syncthreads();
  for (int o = 128; o; o >>= 1) {
    if (t < o) { su[t] = fmaxf(su[t], su[t + o]); sm[t] = fmaxf(sm[t], sm[t + o]); }
    __syncthreads();
  }
  if (t == 0) { mxU[b] = su[0]; mxM[b] = sm[0]; }
}

// --------- per-row softmax stats: m_r (max logit) and rs_r = 1/sum(exp)
// mode 0 (GAT): masked -> -9e15 (exact zero prob), mode 1 (cross): additive -1e4
__global__ void re2g_rowstats(const float* __restrict__ f1, const float* __restrict__ f2,
                              const float* __restrict__ mask, const float* __restrict__ mxU,
                              const float* __restrict__ mxM, float* __restrict__ mrow,
                              float* __restrict__ rsrow, float alpha, int mode) {
  int r = blockIdx.x * 4 + (threadIdx.x >> 5);
  int lane = threadIdx.x & 31;
  if (r >= BSn) return;
  int b = r >> 11;
  float f1v = f1[r];
  float m;
  if (mode == 0) {
    m = lrelu(f1v + mxU[b], alpha);
  } else {
    float a1 = (mxU[b] > -1e30f) ? lrelu(f1v + mxU[b], alpha) : -3.4e38f;
    float a2 = (mxM[b] > -1e30f) ? (lrelu(f1v + mxM[b], alpha) - 10000.f) : -3.4e38f;
    m = fmaxf(a1, a2);
  }
  const float* f2b = f2 + b * Sn;
  const float* mb = mask + b * Sn;
  float s = 0.f;
  for (int j = lane; j < Sn; j += 32) {
    float e = lrelu(f1v + f2b[j], alpha);
    if (mode == 0) {
      if (mb[j] > 0.f) s += __expf(e - m);
    } else {
      s += __expf(e + (1.f - mb[j]) * -10000.f - m);
    }
  }
#pragma unroll
  for (int off = 16; off; off >>= 1) s += __shfl_xor(s, off, 32);
  if (lane == 0) { mrow[r] = m; rsrow[r] = 1.f / s; }
}

// -------- write normalized attention: f32 to d_out region + f16 to workspace
// grid = (Sn/256, BSn)
__global__ void re2g_attwrite(const float* __restrict__ f1, const float* __restrict__ f2,
                              const float* __restrict__ mask, const float* __restrict__ mrow,
                              const float* __restrict__ rsrow, float* __restrict__ Pout,
                              _Float16* __restrict__ Ph, float alpha, int mode) {
  int r = blockIdx.y;
  int j = blockIdx.x * 256 + threadIdx.x;
  int b = r >> 11;
  float e = lrelu(f1[r] + f2[b * Sn + j], alpha);
  float p;
  if (mode == 0) {
    p = (mask[b * Sn + j] > 0.f) ? __expf(e - mrow[r]) * rsrow[r] : 0.f;
  } else {
    p = __expf(e + (1.f - mask[b * Sn + j]) * -10000.f - mrow[r]) * rsrow[r];
  }
  long long idx = (long long)r * Sn + j;
  Pout[idx] = p;
  Ph[idx] = (_Float16)p;
}

// ------- gated fusion: g = sigmoid([reps,ctx].gw); out = reps + g*ctx
__global__ void re2g_gateout(const float* __restrict__ reps, const float* __restrict__ ctx,
                             const float* __restrict__ gw, float* __restrict__ out, int rows) {
  int r = blockIdx.x * 4 + (threadIdx.x >> 5);
  int lane = threadIdx.x & 31;
  if (r >= rows) return;
  const float* rp = reps + (long long)r * Dn;
  const float* cp = ctx + (long long)r * Dn;
  float s = 0.f;
#pragma unroll
  for (int i = 0; i < 8; ++i) {
    int k = lane * 8 + i;
    s += rp[k] * gw[k] + cp[k] * gw[Dn + k];
  }
#pragma unroll
  for (int off = 16; off; off >>= 1) s += __shfl_xor(s, off, 32);  // all lanes get sum
  float g = 1.f / (1.f + __expf(-s));
  float* op = out + (long long)r * Dn;
#pragma unroll
  for (int i = 0; i < 8; ++i) {
    int k = lane * 8 + i;
    op[k] = rp[k] + g * cp[k];
  }
}

extern "C" void kernel_launch(void* const* d_in, const int* in_sizes, int n_in,
                              void* d_out, int out_size, void* d_ws, size_t ws_size,
                              hipStream_t stream) {
  (void)in_sizes; (void)n_in; (void)out_size; (void)ws_size;
  const float* review_in  = (const float*)d_in[0];
  const float* reply_in   = (const float*)d_in[1];
  const float* review_msk = (const float*)d_in[2];
  const float* reply_msk  = (const float*)d_in[3];
  const float* W_rev = (const float*)d_in[4];
  const float* a_rev = (const float*)d_in[5];
  const float* W_rep = (const float*)d_in[6];
  const float* a_rep = (const float*)d_in[7];
  const float* W1 = (const float*)d_in[8];
  const float* a1 = (const float*)d_in[9];
  const float* W2 = (const float*)d_in[10];
  const float* a2 = (const float*)d_in[11];
  const float* gate_rev = (const float*)d_in[12];
  const float* gate_rep = (const float*)d_in[13];
  float* out = (float*)d_out;
  float* out_rev_att = out + 2 * BSDn;
  float* out_rep_att = out_rev_att + BSSn;
  float* out_cross1  = out_rep_att + BSSn;
  float* out_cross2  = out_cross1 + BSSn;

  // ---- workspace bump allocator
  char* wp = (char*)d_ws;
  auto alloc = [&](size_t bytes) { void* r = wp; wp += (bytes + 255) & ~(size_t)255; return r; };
  _Float16* rev_h   = (_Float16*)alloc(BSDn * 2);
  _Float16* rep_h   = (_Float16*)alloc(BSDn * 2);
  _Float16* Wrev_t  = (_Float16*)alloc(Dn * Dn * 2);
  _Float16* Wrep_t  = (_Float16*)alloc(Dn * Dn * 2);
  _Float16* W1_t    = (_Float16*)alloc(Dn * Dn * 2);
  _Float16* W2_t    = (_Float16*)alloc(Dn * Dn * 2);
  _Float16* Whrev_t = (_Float16*)alloc(BSDn * 2);   // [b][D][S]
  _Float16* Whrep_t = (_Float16*)alloc(BSDn * 2);
  _Float16* Wh11_t  = (_Float16*)alloc(BSDn * 2);
  _Float16* Wh12_t  = (_Float16*)alloc(BSDn * 2);
  _Float16* Wh21_t  = (_Float16*)alloc(BSDn * 2);
  _Float16* Wh22_t  = (_Float16*)alloc(BSDn * 2);
  float*    reps_rev = (float*)alloc(BSDn * 4);
  float*    reps_rep = (float*)alloc(BSDn * 4);
  _Float16* reps_rev_h = (_Float16*)alloc(BSDn * 2);  // row-major (GEMM-A use)
  _Float16* reps_rep_h = (_Float16*)alloc(BSDn * 2);
  float*    ctx1 = (float*)alloc(BSDn * 4);
  float*    ctx2 = (float*)alloc(BSDn * 4);
  float *f1v[4], *f2v[4], *mrow[4], *rsrow[4], *mxU[4], *mxM[4];
  for (int i = 0; i < 4; ++i) {
    f1v[i] = (float*)alloc(BSn * 4); f2v[i] = (float*)alloc(BSn * 4);
    mrow[i] = (float*)alloc(BSn * 4); rsrow[i] = (float*)alloc(BSn * 4);
    mxU[i] = (float*)alloc(Bn * 4); mxM[i] = (float*)alloc(Bn * 4);
  }
  _Float16* Ph = (_Float16*)alloc(BSSn * 2);   // reused for all 4 attention mats

  auto gemm = [&](const _Float16* A, const _Float16* BT, const float* resid, float* Cf,
                  _Float16* Ch, _Float16* Cht, int M, int K, int N,
                  long long sA, long long sB, long long sC, int batch) {
    dim3 g(N / 64, M / 64, batch);
    re2g_gemm_wmma<<<g, 128, 0, stream>>>(A, BT, resid, Cf, Ch, Cht, K, N, sA, sB, sC);
  };
  auto attention = [&](const _Float16* WhT_f1, const _Float16* WhT_f2, const float* avec,
                       const float* mask, int slot, float alpha, int mode, float* att_out) {
    re2g_rowdot_t<<<BSn / 256, 256, 0, stream>>>(WhT_f1, avec, f1v[slot]);
    re2g_rowdot_t<<<BSn / 256, 256, 0, stream>>>(WhT_f2, avec + Dn, f2v[slot]);
    re2g_batchmax<<<Bn, 256, 0, stream>>>(f2v[slot], mask, mxU[slot], mxM[slot]);
    re2g_rowstats<<<BSn / 4, 128, 0, stream>>>(f1v[slot], f2v[slot], mask, mxU[slot],
                                               mxM[slot], mrow[slot], rsrow[slot], alpha, mode);
    dim3 gw(Sn / 256, BSn);
    re2g_attwrite<<<gw, 256, 0, stream>>>(f1v[slot], f2v[slot], mask, mrow[slot],
                                          rsrow[slot], att_out, Ph, alpha, mode);
  };
  const long long sAatt = (long long)Sn * Sn;
  const long long sBT   = (long long)Dn * Sn;
  const long long sCC   = (long long)Sn * Dn;

  // ---- convert inputs (row-major f16) and weights (transposed f16)
  re2g_cvt_f16<<<4096, 256, 0, stream>>>(review_in, rev_h, BSDn);
  re2g_cvt_f16<<<4096, 256, 0, stream>>>(reply_in, rep_h, BSDn);
  re2g_cvt_tr<<<256, 256, 0, stream>>>(W_rev, Wrev_t);
  re2g_cvt_tr<<<256, 256, 0, stream>>>(W_rep, Wrep_t);
  re2g_cvt_tr<<<256, 256, 0, stream>>>(W1, W1_t);
  re2g_cvt_tr<<<256, 256, 0, stream>>>(W2, W2_t);

  // ---- GAT review: Wh = h@W ; att = softmax(LR(f1+f2^T)) ; reps = h + att@Wh
  gemm(rev_h, Wrev_t, nullptr, nullptr, nullptr, Whrev_t, BSn, Dn, Dn, 0, 0, 0, 1);
  attention(Whrev_t, Whrev_t, a_rev, review_msk, 0, 0.2f, 0, out_rev_att);
  gemm(Ph, Whrev_t, review_in, reps_rev, reps_rev_h, nullptr, Sn, Sn, Dn, sAatt, sBT, sCC, Bn);

  // ---- GAT reply
  gemm(rep_h, Wrep_t, nullptr, nullptr, nullptr, Whrep_t, BSn, Dn, Dn, 0, 0, 0, 1);
  attention(Whrep_t, Whrep_t, a_rep, reply_msk, 1, 0.2f, 0, out_rep_att);
  gemm(Ph, Whrep_t, reply_in, reps_rep, reps_rep_h, nullptr, Sn, Sn, Dn, sAatt, sBT, sCC, Bn);

  // ---- cross attention 1: p1 = softmax(LR(Wh11.a1lo + (Wh12.a1hi)^T) + rep_inf)
  gemm(reps_rev_h, W1_t, nullptr, nullptr, nullptr, Wh11_t, BSn, Dn, Dn, 0, 0, 0, 1);
  gemm(reps_rep_h, W1_t, nullptr, nullptr, nullptr, Wh12_t, BSn, Dn, Dn, 0, 0, 0, 1);
  attention(Wh11_t, Wh12_t, a1, reply_msk, 2, 0.01f, 1, out_cross1);
  gemm(Ph, Wh12_t, nullptr, ctx1, nullptr, nullptr, Sn, Sn, Dn, sAatt, sBT, sCC, Bn);

  // ---- cross attention 2: p2 = softmax(LR(Wh22.a2lo + (Wh21.a2hi)^T) + rev_inf)
  gemm(reps_rep_h, W2_t, nullptr, nullptr, nullptr, Wh22_t, BSn, Dn, Dn, 0, 0, 0, 1);
  gemm(reps_rev_h, W2_t, nullptr, nullptr, nullptr, Wh21_t, BSn, Dn, Dn, 0, 0, 0, 1);
  attention(Wh22_t, Wh21_t, a2, review_msk, 3, 0.01f, 1, out_cross2);
  gemm(Ph, Wh21_t, nullptr, ctx2, nullptr, nullptr, Sn, Sn, Dn, sAatt, sBT, sCC, Bn);

  // ---- gated outputs
  re2g_gateout<<<BSn / 4, 128, 0, stream>>>(reps_rev, ctx1, gate_rev, out, BSn);
  re2g_gateout<<<BSn / 4, 128, 0, stream>>>(reps_rep, ctx2, gate_rep, out + BSDn, BSn);
}
